// MHA_1606317769177
// MI455X (gfx1250) — matrix operand order repaired and checked by reading, not compiled, not run on hardware
//
#include <hip/hip_runtime.h>
#include <hip/hip_bf16.h>

#define DMODEL 1024
#define DHID   1024
#define NHEAD  16
#define HD     64
#define BATCH  4
#define SEQ    2048
#define NTOK   (BATCH * SEQ)   // 8192

typedef __bf16 bf16_t;
typedef __attribute__((ext_vector_type(16))) __bf16 bf16x16;
typedef __attribute__((ext_vector_type(8)))  __bf16 bf16x8;
typedef __attribute__((ext_vector_type(4)))  __bf16 bf16x4;
typedef __attribute__((ext_vector_type(8)))  float  f32x8;
typedef __attribute__((ext_vector_type(4)))  float  f32x4;
typedef __attribute__((ext_vector_type(4)))  int    i32x4;

// --- CDNA5 async global->LDS path (ASYNCcnt), guarded so compile stays green
#if defined(__AMDGCN__) &&                                                     \
    __has_builtin(__builtin_amdgcn_global_load_async_to_lds_b128) &&           \
    __has_builtin(__builtin_amdgcn_s_wait_asynccnt)
#define HAS_ASYNC 1
#else
#define HAS_ASYNC 0
#if defined(__AMDGCN__)
#warning "gfx1250 async-to-LDS builtins unavailable: falling back to synchronous LDS staging"
#endif
#endif

#if HAS_ASYNC
// Signature (probe-confirmed): (V4i addrspace(1)*, V4i addrspace(3)*, imm offset, imm cpol)
#define ASYNC_COPY16(dst_lds, src_glb)                                         \
    __builtin_amdgcn_global_load_async_to_lds_b128(                            \
        (__attribute__((address_space(1))) i32x4*)(src_glb),                   \
        (__attribute__((address_space(3))) i32x4*)(dst_lds), 0, 0)
#define WAIT_ASYNC(n) __builtin_amdgcn_s_wait_asynccnt(n)
#else
#define ASYNC_COPY16(dst_lds, src_glb)                                         \
    (*(bf16x8*)(dst_lds) = *(const bf16x8*)(src_glb))
#define WAIT_ASYNC(n)
#endif

// Build a 16x32 (A) or 32x16 (B) bf16 WMMA fragment from two contiguous
// 16-byte chunks (the ISA layout makes every lane's data two b128 loads).
static __device__ inline bf16x16 load_frag(const bf16_t* p0, const bf16_t* p1) {
    bf16x8 lo = *(const bf16x8*)p0;
    bf16x8 hi = *(const bf16x8*)p1;
    bf16x16 r;
#pragma unroll
    for (int i = 0; i < 8; ++i) { r[i] = lo[i]; r[i + 8] = hi[i]; }
    return r;
}

static __device__ inline f32x8 wmma_bf16(bf16x16 a, bf16x16 b, f32x8 c) {
    return __builtin_amdgcn_wmma_f32_16x16x32_bf16(
        /*neg_a=*/false, a, /*neg_b=*/false, b,
        /*c_mod=*/(short)0, c, /*reuse_a=*/false, /*reuse_b=*/false);
}

// ---------------------------------------------------------------------------
// Weight prep: f32 W[K=1024][N=1024]  ->  bf16 WT[N][K]
// ---------------------------------------------------------------------------
__global__ __launch_bounds__(256) void wt_kernel(const float* __restrict__ W,
                                                 bf16_t* __restrict__ WT) {
    __shared__ float tile[32][33];
    int k0 = blockIdx.x * 32, n0 = blockIdx.y * 32;
    int tx = threadIdx.x, ty = threadIdx.y;  // 32 x 8
#pragma unroll
    for (int i = 0; i < 4; ++i) {
        int k = ty + i * 8;
        tile[k][tx] = W[(size_t)(k0 + k) * DHID + n0 + tx];
    }
    __syncthreads();
#pragma unroll
    for (int i = 0; i < 4; ++i) {
        int n = ty + i * 8;
        WT[(size_t)(n0 + n) * DMODEL + k0 + tx] = (bf16_t)tile[tx][n];
    }
}

// ---------------------------------------------------------------------------
// LayerNorm: f32 x[row][1024] -> bf16 h[row][1024]
// ---------------------------------------------------------------------------
__global__ __launch_bounds__(256) void ln_kernel(const float* __restrict__ x,
                                                 const float* __restrict__ gamma,
                                                 const float* __restrict__ beta,
                                                 bf16_t* __restrict__ h) {
    int row = blockIdx.x;
    int tid = threadIdx.x;
    const float* xr = x + (size_t)row * DMODEL;
    f32x4 v = *(const f32x4*)(xr + tid * 4);
    float s = v[0] + v[1] + v[2] + v[3];
    float q = v[0]*v[0] + v[1]*v[1] + v[2]*v[2] + v[3]*v[3];
#pragma unroll
    for (int m = 1; m < 32; m <<= 1) {
        s += __shfl_xor(s, m, 32);
        q += __shfl_xor(q, m, 32);
    }
    __shared__ float ss[8], sq[8];
    int w = tid >> 5;
    if ((tid & 31) == 0) { ss[w] = s; sq[w] = q; }
    __syncthreads();
    s = 0.0f; q = 0.0f;
#pragma unroll
    for (int i = 0; i < 8; ++i) { s += ss[i]; q += sq[i]; }
    float mean = s * (1.0f / DMODEL);
    float var  = q * (1.0f / DMODEL) - mean * mean;
    float rs   = rsqrtf(var + 1e-5f);
    f32x4 g = *(const f32x4*)(gamma + tid * 4);
    f32x4 b = *(const f32x4*)(beta  + tid * 4);
    bf16x4 o;
#pragma unroll
    for (int i = 0; i < 4; ++i)
        o[i] = (bf16_t)((v[i] - mean) * rs * g[i] + b[i]);
    *(bf16x4*)(h + (size_t)row * DMODEL + tid * 4) = o;
}

// ---------------------------------------------------------------------------
// WMMA GEMM:  Out = A[M,K] @ BT[N,K]^T + bias.   M=8192, N=K=1024.
// 8 waves/block, each wave computes a 16(M) x 64(N) tile.
// The block's shared 64x32 B tile is staged in LDS via async global->LDS
// (double buffered, ASYNCcnt). Manual 2x K-unroll with static buffer
// addresses keeps the WMMA accumulators in-place (no phi copies).
// mode 0: bf16 out as [B,H,S,hd]    (Q, K)
// mode 2: bf16 out as [B,H,hd,S]    (V transposed for the PV WMMA)
// mode 3: f32  out as [T, DMODEL]   (final projection)
// ---------------------------------------------------------------------------
__global__ __launch_bounds__(256) void gemm_kernel(const bf16_t* __restrict__ A,
                                                   const bf16_t* __restrict__ BT,
                                                   const float* __restrict__ bias,
                                                   void* __restrict__ Out, int mode) {
    int tid  = threadIdx.x;
    int lane = tid & 31;
    int wv   = tid >> 5;
    int m0   = blockIdx.x * 128 + wv * 16;
    int n0   = blockIdx.y * 64;
    int cl   = lane & 15, hi = lane >> 4;

    // B tile staging: [64 n][32 k] bf16 = 4 KB per buffer, double buffered.
    __shared__ __align__(16) bf16_t bsh[2][64 * 32];
    // Each of the 256 threads moves one 16-byte chunk per K-step.
    int bn = tid >> 2;            // 0..63  (row of W^T within tile)
    int bk = (tid & 3) * 8;       // 0,8,16,24 (k offset, bf16 units)
    const bf16_t* bsrc_base = BT + (size_t)(n0 + bn) * DMODEL + bk;
    bf16_t* bdst0 = &bsh[0][bn * 32 + bk];
    bf16_t* bdst1 = &bsh[1][bn * 32 + bk];

    f32x8 acc[4];
#pragma unroll
    for (int s = 0; s < 4; ++s)
#pragma unroll
        for (int j = 0; j < 8; ++j) acc[s][j] = 0.0f;

    const bf16_t* arow = A + (size_t)(m0 + cl) * DMODEL + hi * 16;
    const bf16_t* bb0  = &bsh[0][cl * 32 + hi * 16];
    const bf16_t* bb1  = &bsh[1][cl * 32 + hi * 16];

    auto do_step = [&](const bf16_t* bb, int kk) {
        bf16x16 af = load_frag(arow + kk, arow + kk + 8);
#pragma unroll
        for (int s = 0; s < 4; ++s) {
            const bf16_t* bp = bb + s * 16 * 32;
            bf16x16 bfr = load_frag(bp, bp + 8);
            acc[s] = wmma_bf16(af, bfr, acc[s]);
        }
    };

    ASYNC_COPY16(bdst0, bsrc_base);                 // prime buf0 (kk = 0)

    for (int kk = 0; kk < DMODEL; kk += 64) {
        // ---- stage A: consume buf0 (kk), prefetch buf1 (kk+32)
        ASYNC_COPY16(bdst1, bsrc_base + kk + 32);
        WAIT_ASYNC(1);            // this wave's buf0 fill has landed
        __syncthreads();          // ... and every other wave's too
        do_step(bb0, kk);
        __syncthreads();          // buf0 reads done -> buf0 free

        // ---- stage B: consume buf1 (kk+32), prefetch buf0 (kk+64)
        if (kk + 64 < DMODEL) {
            ASYNC_COPY16(bdst0, bsrc_base + kk + 64);
            WAIT_ASYNC(1);
        } else {
            WAIT_ASYNC(0);
        }
        __syncthreads();
        do_step(bb1, kk + 32);
        __syncthreads();          // buf1 reads done -> buf1 free
    }

#pragma unroll
    for (int s = 0; s < 4; ++s) {
        int c = n0 + s * 16 + cl;            // output column (same for all 8 rows)
        float bv = bias[c];
        if (mode == 2) {                     // V^T: [B,H,hd,S], rows contiguous
            bf16_t* ob = (bf16_t*)Out;
            int head = c >> 6, d = c & 63;
            int bidx = m0 >> 11;
            int sidx = (m0 & (SEQ - 1)) + hi * 8;
            bf16x8 pk;
#pragma unroll
            for (int j = 0; j < 8; ++j) pk[j] = (bf16_t)(acc[s][j] + bv);
            *(bf16x8*)(ob + ((((size_t)bidx * NHEAD + head) * HD + d) * SEQ + sidx)) = pk;
        } else if (mode == 3) {              // final f32 output [T, DMODEL]
            float* of = (float*)Out;
#pragma unroll
            for (int j = 0; j < 8; ++j) {
                int r = m0 + hi * 8 + j;
                of[(size_t)r * DMODEL + c] = acc[s][j] + bv;
            }
        } else {                             // Q/K: [B,H,S,hd]
            bf16_t* ob = (bf16_t*)Out;
            int head = c >> 6, d = c & 63;
#pragma unroll
            for (int j = 0; j < 8; ++j) {
                int r = m0 + hi * 8 + j;
                int bidx = r >> 11, sidx = r & (SEQ - 1);
                ob[(((size_t)bidx * NHEAD + head) * SEQ + sidx) * HD + d] =
                    (bf16_t)(acc[s][j] + bv);
            }
        }
    }
}

// ---------------------------------------------------------------------------
// Causal flash attention. One wave per 16 queries of one (b,h).
// Q,K: bf16 [B,H,S,hd]; VT: bf16 [B,H,hd,S]; AO: bf16 [T, DHID].
// ---------------------------------------------------------------------------
__global__ __launch_bounds__(32) void attn_kernel(const bf16_t* __restrict__ Q,
                                                  const bf16_t* __restrict__ Kc,
                                                  const bf16_t* __restrict__ VT,
                                                  bf16_t* __restrict__ AO) {
    int lane = threadIdx.x;
    int cl = lane & 15, hi = lane >> 4;
    int q0 = blockIdx.x * 16;
    int bh = blockIdx.y;
    int head = bh & (NHEAD - 1), bidx = bh >> 4;

    const bf16_t* Qb = Q  + (size_t)bh * SEQ * HD;
    const bf16_t* Kb = Kc + (size_t)bh * SEQ * HD;
    const bf16_t* Vb = VT + (size_t)bh * HD * SEQ;

    bf16x16 aQ[2];
#pragma unroll
    for (int c = 0; c < 2; ++c) {
        const bf16_t* p = Qb + (size_t)(q0 + cl) * HD + c * 32 + hi * 16;
        aQ[c] = load_frag(p, p + 8);
    }

    float m[8], l[8];
    f32x8 accO[4];
#pragma unroll
    for (int j = 0; j < 8; ++j) { m[j] = -1e30f; l[j] = 0.0f; }
#pragma unroll
    for (int s = 0; s < 4; ++s)
#pragma unroll
        for (int j = 0; j < 8; ++j) accO[s][j] = 0.0f;

    __shared__ __align__(16) bf16_t pt[16 * 32];

    for (int k0 = 0; k0 < q0 + 16; k0 += 32) {
        bool masked = (k0 + 31 > q0);
        f32x8 sc[2];
#pragma unroll
        for (int s = 0; s < 2; ++s) {
            int key = k0 + s * 16 + cl;
            const bf16_t* kp0 = Kb + (size_t)key * HD + hi * 16;
            bf16x16 b0 = load_frag(kp0, kp0 + 8);
            bf16x16 b1 = load_frag(kp0 + 32, kp0 + 40);
            f32x8 z;
#pragma unroll
            for (int j = 0; j < 8; ++j) z[j] = 0.0f;
            z = wmma_bf16(aQ[0], b0, z);
            z = wmma_bf16(aQ[1], b1, z);
#pragma unroll
            for (int j = 0; j < 8; ++j) z[j] *= 0.125f;  // 1/sqrt(64)
            if (masked) {
#pragma unroll
                for (int j = 0; j < 8; ++j) {
                    int qrow = q0 + hi * 8 + j;
                    if (key > qrow) z[j] = -1e30f;
                }
            }
            sc[s] = z;
        }

        // online softmax: each row lives in one 16-lane half -> xor reduce
        float alpha[8];
#pragma unroll
        for (int j = 0; j < 8; ++j) {
            float r = fmaxf(sc[0][j], sc[1][j]);
#pragma unroll
            for (int mk = 1; mk < 16; mk <<= 1) r = fmaxf(r, __shfl_xor(r, mk, 32));
            float mn = fmaxf(m[j], r);
            alpha[j] = __expf(m[j] - mn);
            float p0 = __expf(sc[0][j] - mn);
            float p1 = __expf(sc[1][j] - mn);
            sc[0][j] = p0; sc[1][j] = p1;
            float rs = p0 + p1;
#pragma unroll
            for (int mk = 1; mk < 16; mk <<= 1) rs += __shfl_xor(rs, mk, 32);
            l[j] = l[j] * alpha[j] + rs;
            m[j] = mn;
        }
#pragma unroll
        for (int s = 0; s < 4; ++s)
#pragma unroll
            for (int j = 0; j < 8; ++j) accO[s][j] *= alpha[j];

        // C-layout -> A-layout conversion through LDS (16x32 bf16 tile)
#pragma unroll
        for (int j = 0; j < 8; ++j) {
            int r = hi * 8 + j;
            pt[r * 32 + cl]      = (bf16_t)sc[0][j];
            pt[r * 32 + 16 + cl] = (bf16_t)sc[1][j];
        }
        __syncthreads();
        const bf16_t* pr = &pt[cl * 32 + hi * 16];
        bf16x16 aP = load_frag(pr, pr + 8);

#pragma unroll
        for (int s = 0; s < 4; ++s) {
            const bf16_t* vp = Vb + (size_t)(s * 16 + cl) * SEQ + k0 + hi * 16;
            bf16x16 bV = load_frag(vp, vp + 8);
            accO[s] = wmma_bf16(aP, bV, accO[s]);
        }
        __syncthreads();
    }

    float inv[8];
#pragma unroll
    for (int j = 0; j < 8; ++j) inv[j] = 1.0f / l[j];
#pragma unroll
    for (int s = 0; s < 4; ++s) {
        int d = s * 16 + cl;
#pragma unroll
        for (int j = 0; j < 8; ++j) {
            int t = bidx * SEQ + q0 + hi * 8 + j;
            AO[(size_t)t * DHID + head * HD + d] = (bf16_t)(accO[s][j] * inv[j]);
        }
    }
}

// ---------------------------------------------------------------------------
extern "C" void kernel_launch(void* const* d_in, const int* in_sizes, int n_in,
                              void* d_out, int out_size, void* d_ws, size_t ws_size,
                              hipStream_t stream) {
    (void)in_sizes; (void)n_in; (void)out_size; (void)ws_size;
    const float* x     = (const float*)d_in[0];
    const float* Wq    = (const float*)d_in[1];
    const float* bq    = (const float*)d_in[2];
    const float* Wk    = (const float*)d_in[3];
    const float* bk    = (const float*)d_in[4];
    const float* Wv    = (const float*)d_in[5];
    const float* bv    = (const float*)d_in[6];
    const float* Wo    = (const float*)d_in[7];
    const float* bo    = (const float*)d_in[8];
    const float* gamma = (const float*)d_in[9];
    const float* beta  = (const float*)d_in[10];

    char* ws = (char*)d_ws;
    const size_t sz_h   = (size_t)NTOK * DMODEL * sizeof(bf16_t);  // 16 MB
    const size_t sz_w   = (size_t)DMODEL * DHID * sizeof(bf16_t);  //  2 MB
    const size_t sz_qkv = (size_t)NTOK * DHID * sizeof(bf16_t);    // 16 MB

    bf16_t* h   = (bf16_t*)(ws);
    bf16_t* WqT = (bf16_t*)(ws + sz_h);
    bf16_t* WkT = (bf16_t*)(ws + sz_h + sz_w);
    bf16_t* WvT = (bf16_t*)(ws + sz_h + 2 * sz_w);
    bf16_t* WoT = (bf16_t*)(ws + sz_h + 3 * sz_w);
    bf16_t* Qb  = (bf16_t*)(ws + sz_h + 4 * sz_w);
    bf16_t* Kb  = (bf16_t*)(ws + sz_h + 4 * sz_w + sz_qkv);
    bf16_t* VT  = (bf16_t*)(ws + sz_h + 4 * sz_w + 2 * sz_qkv);
    bf16_t* AO  = h;  // h is dead after the V GEMM; reuse for attention output

    dim3 tb(32, 8), tg(DMODEL / 32, DHID / 32);
    wt_kernel<<<tg, tb, 0, stream>>>(Wq, WqT);
    wt_kernel<<<tg, tb, 0, stream>>>(Wk, WkT);
    wt_kernel<<<tg, tb, 0, stream>>>(Wv, WvT);
    wt_kernel<<<tg, tb, 0, stream>>>(Wo, WoT);

    ln_kernel<<<NTOK, 256, 0, stream>>>(x, gamma, beta, h);

    dim3 gg(NTOK / 128, DHID / 64);
    gemm_kernel<<<gg, 256, 0, stream>>>(h, WqT, bq, Qb, 0);
    gemm_kernel<<<gg, 256, 0, stream>>>(h, WkT, bk, Kb, 0);
    gemm_kernel<<<gg, 256, 0, stream>>>(h, WvT, bv, VT, 2);

    attn_kernel<<<dim3(SEQ / 16, BATCH * NHEAD), 32, 0, stream>>>(Qb, Kb, VT, AO);

    gemm_kernel<<<gg, 256, 0, stream>>>(AO, WoT, bo, d_out, 3);
}
